// BotRGCN2_5531917877298
// MI455X (gfx1250) — matrix operand order repaired and checked by reading, not compiled, not run on hardware
//
#include <hip/hip_runtime.h>

typedef __attribute__((ext_vector_type(16))) _Float16 v16h;
typedef __attribute__((ext_vector_type(8)))  _Float16 v8h;
typedef __attribute__((ext_vector_type(2)))  _Float16 v2h;
typedef __attribute__((ext_vector_type(8)))  float    v8f;

// ---------------------------------------------------------------------------
// Generic fused GEMM:  C[N,128] = act( A[N,K] @ W[K,128] + bias )
// f16 WMMA (v_wmma_f32_16x16x32_f16) with fp32 accumulation.
// Block: 256 threads (8 wave32), 64 rows x 128 cols of C per block.
// Wave w owns column tile w (16 cols) across 4 row tiles.
// LDS pad stride 40 halves (80 B): 16B-aligned rows for b128 stores,
// 20*m mod 64 distinct for m=0..15 -> conflict-free fragment reads.
// ---------------------------------------------------------------------------
#define LDS_STRIDE 40

__global__ void __launch_bounds__(256)
gemm_wmma_kernel(const float* __restrict__ A, const float* __restrict__ W,
                 const float* __restrict__ bias, float* __restrict__ C,
                 int N, int K, int do_lrelu) {
    __shared__ __align__(16) _Float16 lgA[64 * LDS_STRIDE];   // A chunk  [row][k], 64x32
    __shared__ __align__(16) _Float16 lgW[128 * LDS_STRIDE];  // W chunk transposed [col][k]

    const int tid     = threadIdx.x;
    const int wave    = tid >> 5;
    const int lane    = tid & 31;
    const int half_id = lane >> 4;       // 0: lanes 0-15, 1: lanes 16-31
    const int l16     = lane & 15;
    const int col     = wave * 16 + l16; // output column owned by this lane
    const int blockRow = blockIdx.x * 64;

    // A staging ownership: thread -> (row 0..63, k-octet). Clamp row so loads are
    // unconditional; garbage rows only feed output rows that are never stored.
    const int ar   = tid >> 2;
    const int akq  = (tid & 3) * 8;
    int arow = blockRow + ar;
    if (arow > N - 1) arow = N - 1;
    const float* aptr = A + (size_t)arow * K + akq;

    v8f acc[4];
#pragma unroll
    for (int i = 0; i < 4; ++i) acc[i] = (v8f)(0.0f);

    for (int kk = 0; kk < K; kk += 32) {
        // ---- stage A: 8 contiguous floats/thread -> 2x b128 load, 1x b128 store
        {
            const float4* ap = (const float4*)(aptr + kk);
            float4 a0 = ap[0];
            float4 a1 = ap[1];
            v8h ha;
            ha[0] = (_Float16)a0.x; ha[1] = (_Float16)a0.y;
            ha[2] = (_Float16)a0.z; ha[3] = (_Float16)a0.w;
            ha[4] = (_Float16)a1.x; ha[5] = (_Float16)a1.y;
            ha[6] = (_Float16)a1.z; ha[7] = (_Float16)a1.w;
            *(v8h*)(&lgA[ar * LDS_STRIDE + akq]) = ha;
        }
        // ---- stage W transposed: unit = (k-pair k2, col-quad nq); coalesced
        //      float4 loads from rows k=2*k2 and 2*k2+1, packed dword LDS stores.
#pragma unroll
        for (int p = 0; p < 2; ++p) {
            int u  = tid + 256 * p;          // 0..511
            int k2 = u >> 5;                 // 0..15
            int nq = u & 31;                 // 0..31
            const float* wp = W + (size_t)(kk + 2 * k2) * 128 + nq * 4;
            float4 lo = *(const float4*)wp;
            float4 hi = *(const float4*)(wp + 128);
            float lof[4] = {lo.x, lo.y, lo.z, lo.w};
            float hif[4] = {hi.x, hi.y, hi.z, hi.w};
#pragma unroll
            for (int c = 0; c < 4; ++c) {
                v2h pk;
                pk[0] = (_Float16)lof[c];
                pk[1] = (_Float16)hif[c];
                *(v2h*)(&lgW[(nq * 4 + c) * LDS_STRIDE + 2 * k2]) = pk;
            }
        }
        __syncthreads();

        // ---- B fragment (32x16 f16): lane=col; lanes 0-15: K=2vv,2vv+1,
        //      lanes 16-31: K=16+2vv,16+2vv+1  (contiguous -> 2x ds_load_b128)
        v16h bfrag;
#pragma unroll
        for (int vv = 0; vv < 8; ++vv) {
            int k = half_id * 16 + 2 * vv;
            bfrag[2 * vv]     = lgW[col * LDS_STRIDE + k];
            bfrag[2 * vv + 1] = lgW[col * LDS_STRIDE + k + 1];
        }

        // ---- 4 row tiles: A fragment in contiguous K order (matches B), then WMMA
#pragma unroll
        for (int rt = 0; rt < 4; ++rt) {
            v16h afrag;
#pragma unroll
            for (int vv = 0; vv < 8; ++vv) {
                int k = half_id * 16 + 2 * vv;
                afrag[2 * vv]     = lgA[(rt * 16 + l16) * LDS_STRIDE + k];
                afrag[2 * vv + 1] = lgA[(rt * 16 + l16) * LDS_STRIDE + k + 1];
            }
            acc[rt] = __builtin_amdgcn_wmma_f32_16x16x32_f16(
                false, afrag, false, bfrag, (short)0, acc[rt], false, false);
        }
        __syncthreads();
    }

    // ---- epilogue: D layout -> lane col fixed, VGPR r -> row half_id*8 + r.
    // Fast path (uniform branch): whole 64-row block in bounds -> unguarded stores.
    float bv = bias ? bias[col] : 0.0f;
    float* cptr = C + (size_t)(blockRow + half_id * 8) * 128 + col;
    if (blockRow + 64 <= N) {
#pragma unroll
        for (int rt = 0; rt < 4; ++rt) {
#pragma unroll
            for (int r = 0; r < 8; ++r) {
                float val = acc[rt][r] + bv;
                if (do_lrelu) val = (val > 0.0f) ? val : 0.01f * val;
                cptr[(size_t)(rt * 16 + r) * 128] = val;
            }
        }
    } else {
#pragma unroll
        for (int rt = 0; rt < 4; ++rt) {
#pragma unroll
            for (int r = 0; r < 8; ++r) {
                int row = blockRow + rt * 16 + half_id * 8 + r;
                if (row < N) {
                    float val = acc[rt][r] + bv;
                    if (do_lrelu) val = (val > 0.0f) ? val : 0.01f * val;
                    C[(size_t)row * 128 + col] = val;
                }
            }
        }
    }
}

// ---------------------------------------------------------------------------
// Helpers for the RGCN mean aggregation
// ---------------------------------------------------------------------------
__global__ void zero_kernel(float* p, int n) {
    int i = blockIdx.x * 256 + threadIdx.x;
    if (i < n) p[i] = 0.0f;
}

__global__ void count_kernel(const int* __restrict__ ei, const int* __restrict__ et,
                             float* __restrict__ cnt, int E) {
    int e = blockIdx.x * 256 + threadIdx.x;
    if (e < E) {
        int dst = ei[E + e];
        int r   = et[e];
        atomicAdd(&cnt[dst * 2 + r], 1.0f);
    }
}

__global__ void recip_kernel(float* p, int n) {
    int i = blockIdx.x * 256 + threadIdx.x;
    if (i < n) p[i] = 1.0f / fmaxf(p[i], 1.0f);
}

// one wave per edge: agg[dst] += xw[etype][src] * inv_cnt[dst, etype]
// agg (51.2 MB) is L2-resident on MI455X (192 MB L2) -> atomics stay on-chip.
__global__ void __launch_bounds__(256)
scatter_kernel(const float* __restrict__ xw, const float* __restrict__ inv,
               float* __restrict__ agg, const int* __restrict__ ei,
               const int* __restrict__ et, int E, int N) {
    int e    = (int)((blockIdx.x * 256 + threadIdx.x) >> 5);
    int lane = threadIdx.x & 31;
    if (e >= E) return;
    int src = ei[e];
    int dst = ei[E + e];
    int r   = et[e];
    float s = inv[dst * 2 + r];
    const float4 v = *(const float4*)(xw + ((size_t)r * (size_t)N + (size_t)src) * 128 + lane * 4);
    float* a = agg + (size_t)dst * 128 + lane * 4;
    atomicAdd(a + 0, v.x * s);
    atomicAdd(a + 1, v.y * s);
    atomicAdd(a + 2, v.z * s);
    atomicAdd(a + 3, v.w * s);
}

// ---------------------------------------------------------------------------
// out[N,2] = y[N,128] @ W2[128,2] + b2 ; 16 lanes per row, shfl reduction
// ---------------------------------------------------------------------------
__global__ void __launch_bounds__(256)
out2_kernel(const float* __restrict__ y, const float* __restrict__ W2,
            const float* __restrict__ b2, float* __restrict__ out, int N) {
    __shared__ float sW[256];
    __shared__ float sb[2];
    int tid = threadIdx.x;
    sW[tid] = W2[tid];
    if (tid < 2) sb[tid] = b2[tid];
    __syncthreads();

    int node = blockIdx.x * 16 + (tid >> 4);
    int l    = tid & 15;
    float a0 = 0.0f, a1 = 0.0f;
    if (node < N) {
        const float4* row = (const float4*)(y + (size_t)node * 128);
#pragma unroll
        for (int j = 0; j < 2; ++j) {
            float4 v = row[l * 2 + j];
            int k = l * 8 + j * 4;
            a0 += v.x * sW[(k + 0) * 2] + v.y * sW[(k + 1) * 2] +
                  v.z * sW[(k + 2) * 2] + v.w * sW[(k + 3) * 2];
            a1 += v.x * sW[(k + 0) * 2 + 1] + v.y * sW[(k + 1) * 2 + 1] +
                  v.z * sW[(k + 2) * 2 + 1] + v.w * sW[(k + 3) * 2 + 1];
        }
    }
#pragma unroll
    for (int off = 8; off >= 1; off >>= 1) {
        a0 += __shfl_xor(a0, off, 32);
        a1 += __shfl_xor(a1, off, 32);
    }
    if (l == 0 && node < N) {
        out[node * 2 + 0] = a0 + sb[0];
        out[node * 2 + 1] = a1 + sb[1];
    }
}

// ---------------------------------------------------------------------------
extern "C" void kernel_launch(void* const* d_in, const int* in_sizes, int n_in,
                              void* d_out, int out_size, void* d_ws, size_t ws_size,
                              hipStream_t stream) {
    const float* tweet     = (const float*)d_in[1];
    const int*   ei        = (const int*)d_in[4];   // [2,E]
    const int*   et        = (const int*)d_in[5];   // [E]
    const float* W_tweet   = (const float*)d_in[6];
    const float* b_tweet   = (const float*)d_in[7];
    const float* W_in      = (const float*)d_in[8];
    const float* b_in      = (const float*)d_in[9];
    const float* rgcn_w    = (const float*)d_in[10]; // [2,128,128]
    const float* rgcn_root = (const float*)d_in[11];
    const float* rgcn_b    = (const float*)d_in[12];
    const float* W_out1    = (const float*)d_in[13];
    const float* b_out1    = (const float*)d_in[14];
    const float* W_out2    = (const float*)d_in[15];
    const float* b_out2    = (const float*)d_in[16];
    float* out = (float*)d_out;

    const int TW = 768, D = 128;
    const int N = in_sizes[1] / TW;
    const int E = in_sizes[5];

    float* ws = (float*)d_ws;
    size_t ND = (size_t)N * D;
    float* bx  = ws;            // x, later h2
    float* b1  = ws + ND;       // t, later xw0 / hw0
    float* b2  = ws + 2 * ND;   // xw1 / hw1  (b1+ND => relation stride = ND)
    float* bh  = ws + 3 * ND;   // h (conv1 out), later y
    float* cnt = ws + 4 * ND;   // [N,2] counts -> reciprocals

    dim3 blk(256);
    const int gemm_blocks = (N + 63) / 64;
    const int nr = N * 2;

    // per-(dst,relation) reciprocal edge counts (same for both conv layers)
    zero_kernel<<<(nr + 255) / 256, blk, 0, stream>>>(cnt, nr);
    count_kernel<<<(E + 255) / 256, blk, 0, stream>>>(ei, et, cnt, E);
    recip_kernel<<<(nr + 255) / 256, blk, 0, stream>>>(cnt, nr);

    // t = lrelu(tweet @ W_tweet + b_tweet) ; x = lrelu(t @ W_in + b_in)
    gemm_wmma_kernel<<<gemm_blocks, blk, 0, stream>>>(tweet, W_tweet, b_tweet, b1, N, TW, 1);
    gemm_wmma_kernel<<<gemm_blocks, blk, 0, stream>>>(b1, W_in, b_in, bx, N, D, 1);

    // conv1: xw_r = x @ W_r ; h = x @ root + bias ; h += scatter(mean msgs)
    gemm_wmma_kernel<<<gemm_blocks, blk, 0, stream>>>(bx, rgcn_w,         nullptr, b1, N, D, 0);
    gemm_wmma_kernel<<<gemm_blocks, blk, 0, stream>>>(bx, rgcn_w + D * D, nullptr, b2, N, D, 0);
    gemm_wmma_kernel<<<gemm_blocks, blk, 0, stream>>>(bx, rgcn_root,      rgcn_b,  bh, N, D, 0);
    scatter_kernel<<<(E + 7) / 8, blk, 0, stream>>>(b1, cnt, bh, ei, et, E, N);

    // conv2 on h -> h2 (reuses bx; x is dead)
    gemm_wmma_kernel<<<gemm_blocks, blk, 0, stream>>>(bh, rgcn_w,         nullptr, b1, N, D, 0);
    gemm_wmma_kernel<<<gemm_blocks, blk, 0, stream>>>(bh, rgcn_w + D * D, nullptr, b2, N, D, 0);
    gemm_wmma_kernel<<<gemm_blocks, blk, 0, stream>>>(bh, rgcn_root,      rgcn_b,  bx, N, D, 0);
    scatter_kernel<<<(E + 7) / 8, blk, 0, stream>>>(b1, cnt, bx, ei, et, E, N);

    // y = lrelu(h2 @ W_out1 + b_out1) ; out = y @ W_out2 + b_out2
    gemm_wmma_kernel<<<gemm_blocks, blk, 0, stream>>>(bx, W_out1, b_out1, bh, N, D, 1);
    out2_kernel<<<(N + 15) / 16, blk, 0, stream>>>(bh, W_out2, b_out2, out, N);
}
